// QuantizedLinear_84610855731324
// MI455X (gfx1250) — matrix operand order repaired and checked by reading, not compiled, not run on hardware
//
#include <hip/hip_runtime.h>

typedef __attribute__((ext_vector_type(16))) _Float16 v16h;
typedef __attribute__((ext_vector_type(8)))  _Float16 v8h;
typedef __attribute__((ext_vector_type(4)))  _Float16 v4h;
typedef __attribute__((ext_vector_type(8)))  float    v8f;

#define IN_F    8192
#define OUT_F   14336
#define M_TOK   64
#define KSTEP   32
#define NST     4            // weight register-ring depth (consumed 4 sub-steps old)
#define KCHUNK  128          // K elements staged in LDS per chunk
#define NCHUNK  (IN_F / KCHUNK)          // 64
#define LDS_ROW (KCHUNK + 8)             // +8 halfs pad -> 272B row stride, no bank conflicts
#define PF_AHEAD_INT 256     // prefetch 1 KB ahead per weight row

// ---------------------------------------------------------------------------
// Kernel 1: fp32 -> f16 activation convert (int4 weights are exact in f16).
// ---------------------------------------------------------------------------
__global__ __launch_bounds__(256) void xcvt_f16_kernel(const float* __restrict__ x,
                                                       _Float16* __restrict__ xh) {
    const int i = (blockIdx.x * 256 + threadIdx.x) * 4;
    float4 v = *(const float4*)(x + i);
    v4h h;
    h[0] = (_Float16)v.x;
    h[1] = (_Float16)v.y;
    h[2] = (_Float16)v.z;
    h[3] = (_Float16)v.w;
    *(v4h*)(xh + i) = h;
}

// ---- raw CDNA5 sync helpers (no hidden loadcnt drains) ---------------------
__device__ __forceinline__ void wg_barrier() {
    asm volatile("s_barrier_signal -1\n\ts_barrier_wait -1" ::: "memory");
}
__device__ __forceinline__ void wait_async_le8() {
    asm volatile("s_wait_asynccnt 0x8" ::: "memory");
}
__device__ __forceinline__ void wait_async_0() {
    asm volatile("s_wait_asynccnt 0x0" ::: "memory");
}

// ---- async global -> LDS copy, 16B per lane, ASYNCcnt-tracked --------------
__device__ __forceinline__ void async_cp16(uint32_t lds_off, const _Float16* g, int imm) {
    asm volatile("global_load_async_to_lds_b128 %0, %1, off offset:%2"
                 :: "v"(lds_off), "v"(g), "n"(imm) : "memory");
}

// Stage one 64 x KCHUNK f16 chunk of x into an LDS buffer (whole block).
// Thread t: row = t/2, 128B half-row; 8 async b128, same offset both sides.
__device__ __forceinline__ void copy_chunk(_Float16* dstbuf,
                                           const _Float16* __restrict__ xh,
                                           int kbase, int tid) {
    const int crow  = tid >> 1;
    const int chalf = tid & 1;
    const _Float16* g = xh + (size_t)crow * IN_F + kbase + chalf * 64;
    const uint32_t  l = (uint32_t)(uintptr_t)(dstbuf + crow * LDS_ROW + chalf * 64);
    #pragma unroll
    for (int s = 0; s < 8; ++s)
        async_cp16(l, g, s * 16);
}

// ---- A-fragment from LDS: ISA 16-bit A 16x32 layout (ds_load_b128 x2) ------
__device__ __forceinline__ v16h lds_a_frag(const _Float16* sb, int mrow, int hi, int kl) {
    const _Float16* p = sb + mrow * LDS_ROW + kl + hi * 8;
    union { v16h v; v8h h[2]; } a;
    a.h[0] = *(const v8h*)(p);        // K = kl + hi*8 + 0..7
    a.h[1] = *(const v8h*)(p + 16);   // K = kl + 16 + hi*8 + 0..7
    return a.v;
}

// ---- weight K-step load (2x global_load_b128, LOADcnt stream) --------------
__device__ __forceinline__ void load_w(const int* __restrict__ wrow, int hi, int k0,
                                       int4& p0, int4& p1) {
    const int pbase = (k0 >> 1) + hi * 4;
    p0 = *(const int4*)(wrow + pbase);
    p1 = *(const int4*)(wrow + pbase + 8);
}

// Dequant 16 int4 -> v16h B fragment (row scale factored out of the K-sum).
__device__ __forceinline__ v16h dequant16(int4 p0, int4 p1) {
    int pk[8] = {p0.x, p0.y, p0.z, p0.w, p1.x, p1.y, p1.z, p1.w};
    v16h b;
    #pragma unroll
    for (int j = 0; j < 8; ++j) {
        int lo = pk[j] & 0xF;
        int hn = (pk[j] >> 4) & 0xF;
        b[2 * j]     = (_Float16)(short)((lo ^ 8) - 8);
        b[2 * j + 1] = (_Float16)(short)((hn ^ 8) - 8);
    }
    return b;
}

// ---------------------------------------------------------------------------
// Kernel 2: GEMM. Wave = 16 output columns x all 64 rows (4 M-tiles).
// Three decoupled pipelines: ASYNCcnt (x -> LDS), LOADcnt (weight HBM stream,
// 4-deep register ring), DScnt (LDS -> WMMA A-fragments).
// ---------------------------------------------------------------------------
__global__ __launch_bounds__(128) void qlin_int4_wmma_kernel(
    const _Float16* __restrict__ xh,
    const int*      __restrict__ wp,      // [OUT_F][IN_F/2] one packed byte per int32
    const float*    __restrict__ scales,  // [OUT_F]
    const float*    __restrict__ bias,    // [OUT_F]
    float*          __restrict__ out)     // [M_TOK][OUT_F]
{
    __shared__ _Float16 sbuf[2][M_TOK * LDS_ROW];   // 2 x 17 KB

    const int tid   = threadIdx.x;
    const int lane  = tid & 31;
    const int wave  = tid >> 5;
    const int lhalf = lane & 15;
    const int hi    = lane >> 4;

    const int n0  = blockIdx.x * 64 + wave * 16;
    const int col = n0 + lhalf;

    const int* __restrict__ wrow = wp + (size_t)col * (IN_F / 2);

    v8f acc0 = {}, acc1 = {}, acc2 = {}, acc3 = {};

    // Stage chunk 0 of x; prime the weight ring with K-steps 0,32,64,96.
    copy_chunk(&sbuf[0][0], xh, 0, tid);
    int4 wq0[NST], wq1[NST];
    #pragma unroll
    for (int s = 0; s < NST; ++s)
        load_w(wrow, hi, s * KSTEP, wq0[s], wq1[s]);

    for (int c = 0; c < NCHUNK; ++c) {
        // Issue next chunk's async copies, then wait for *this* chunk's
        // (ASYNCcnt completes in order -> <=8 means chunk c landed).
        if (c + 1 < NCHUNK) {
            copy_chunk(&sbuf[(c + 1) & 1][0], xh, (c + 1) * KCHUNK, tid);
            wait_async_le8();
        } else {
            wait_async_0();
        }
        wg_barrier();                     // all waves' copies visible

        const _Float16* cb = &sbuf[c & 1][0];

        #pragma unroll
        for (int js = 0; js < NST; ++js) {          // 4 K-steps per chunk
            const int kl = js * KSTEP;
            const int k  = c * KCHUNK + kl;

            // A fragments from LDS (DScnt only -- never drains LOADcnt)
            v16h a0 = lds_a_frag(cb,  0 + lhalf, hi, kl);
            v16h a1 = lds_a_frag(cb, 16 + lhalf, hi, kl);
            v16h a2 = lds_a_frag(cb, 32 + lhalf, hi, kl);
            v16h a3 = lds_a_frag(cb, 48 + lhalf, hi, kl);

            // Dequant the 4-step-old weight stage, then refill it 128 K ahead.
            v16h b = dequant16(wq0[js], wq1[js]);
            load_w(wrow, hi, (k + NST * KSTEP) & (IN_F - 1), wq0[js], wq1[js]);
            if ((js & 1) == 0)
                __builtin_prefetch(wrow + (k >> 1) + PF_AHEAD_INT, 0, 0);

            acc0 = __builtin_amdgcn_wmma_f32_16x16x32_f16(false, a0, false, b,
                                                          (short)0, acc0, false, false);
            acc1 = __builtin_amdgcn_wmma_f32_16x16x32_f16(false, a1, false, b,
                                                          (short)0, acc1, false, false);
            acc2 = __builtin_amdgcn_wmma_f32_16x16x32_f16(false, a2, false, b,
                                                          (short)0, acc2, false, false);
            acc3 = __builtin_amdgcn_wmma_f32_16x16x32_f16(false, a3, false, b,
                                                          (short)0, acc3, false, false);
        }

        wg_barrier();   // reads of cb done before it is refilled two chunks on
    }

    // ---- Epilogue: y = acc * scale[col] + bias[col]
    const float s  = scales[col];
    const float bb = bias[col];
    const v8f accs[4] = {acc0, acc1, acc2, acc3};
    #pragma unroll
    for (int mt = 0; mt < 4; ++mt) {
        #pragma unroll
        for (int r = 0; r < 8; ++r) {
            const int m = mt * 16 + r + hi * 8;
            out[(size_t)m * OUT_F + col] = accs[mt][r] * s + bb;
        }
    }
}

extern "C" void kernel_launch(void* const* d_in, const int* in_sizes, int n_in,
                              void* d_out, int out_size, void* d_ws, size_t ws_size,
                              hipStream_t stream) {
    (void)in_sizes; (void)n_in; (void)out_size; (void)ws_size;

    const float* x      = (const float*)d_in[0];
    const int*   wp     = (const int*)d_in[1];
    const float* scales = (const float*)d_in[2];
    const float* bias   = (const float*)d_in[3];
    float*       out    = (float*)d_out;
    _Float16*    xh     = (_Float16*)d_ws;   // 1 MB scratch

    const int nconv = (M_TOK * IN_F) / 4;
    hipLaunchKernelGGL(xcvt_f16_kernel, dim3(nconv / 256), dim3(256), 0, stream,
                       x, xh);

    hipLaunchKernelGGL(qlin_int4_wmma_kernel, dim3(OUT_F / 64), dim3(128), 0, stream,
                       xh, wp, scales, bias, out);
}
